// MLP_FP8_13211319402695
// MI455X (gfx1250) — 8-bit matrix operand order repaired and checked by reading, compiled, not run on hardware
//
#include <hip/hip_runtime.h>
#include <stdint.h>


typedef __attribute__((ext_vector_type(16))) int   v16i;
typedef __attribute__((ext_vector_type(8)))  float v8f;
typedef __attribute__((ext_vector_type(4)))  int   v4i;

#define BM 128
#define BN 128
#define BK 128

#if defined(__gfx1250__) && __has_builtin(__builtin_amdgcn_global_load_async_to_lds_b128)
#define ASYNC_LDS 1
typedef __attribute__((address_space(1))) v4i* as1_v4i_ptr;
typedef __attribute__((address_space(3))) v4i* as3_v4i_ptr;
#endif

__device__ __forceinline__ void copy16_to_lds(const unsigned char* g, unsigned char* l) {
#ifdef ASYNC_LDS
  __builtin_amdgcn_global_load_async_to_lds_b128(
      (as1_v4i_ptr)(uintptr_t)g,
      (as3_v4i_ptr)(unsigned int)(uintptr_t)l,
      0, 0);
#else
  *(uint4*)l = *(const uint4*)g;
#endif
}

__device__ __forceinline__ void wait_lds_copies() {
#ifdef ASYNC_LDS
#if __has_builtin(__builtin_amdgcn_s_wait_asynccnt)
  __builtin_amdgcn_s_wait_asynccnt(0);
#else
  asm volatile("s_wait_asynccnt 0x0" ::: "memory");
#endif
#endif
}

__device__ __forceinline__ unsigned char quant1(float v) {
#if __has_builtin(__builtin_amdgcn_cvt_pk_fp8_f32)
  int p = __builtin_amdgcn_cvt_pk_fp8_f32(v, v, 0, false);
  return (unsigned char)(p & 0xff);
#else
  unsigned int u = __float_as_uint(v);
  unsigned char s = (unsigned char)((u >> 24) & 0x80);
  unsigned int a = u & 0x7fffffffu;
  if (a >= 0x7f800000u) return (unsigned char)(s | 0x7f);
  float af = __uint_as_float(a);
  if (af >= 464.0f) return (unsigned char)(s | 0x7e);
  if (af < 0.015625f) { int q = (int)(af * 512.0f + 0.5f); return (unsigned char)(s | (unsigned char)q); }
  unsigned int exp = a >> 23, man = a & 0x7fffffu;
  unsigned int m = man >> 20, rem = man & 0xfffffu;
  if (rem > 0x80000u || (rem == 0x80000u && (m & 1u))) { m++; if (m == 8u) { m = 0u; exp++; } }
  int e8 = (int)exp - 120;
  if (e8 >= 16) return (unsigned char)(s | 0x7e);
  if (e8 <= 0) { int q = (int)(af * 512.0f + 0.5f); return (unsigned char)(s | (unsigned char)q); }
  return (unsigned char)(s | (unsigned char)((e8 << 3) | m));
#endif
}

__device__ __forceinline__ unsigned int quant4(float x, float y, float z, float w) {
#if __has_builtin(__builtin_amdgcn_cvt_pk_fp8_f32)
  int p = 0;
  p = __builtin_amdgcn_cvt_pk_fp8_f32(x, y, p, false);
  p = __builtin_amdgcn_cvt_pk_fp8_f32(z, w, p, true);
  return (unsigned int)p;
#else
  return (unsigned int)quant1(x)        | ((unsigned int)quant1(y) << 8) |
         ((unsigned int)quant1(z) << 16) | ((unsigned int)quant1(w) << 24);
#endif
}

__global__ void __launch_bounds__(256)
quantize_fp8_kernel(const float* __restrict__ in, unsigned char* __restrict__ out, int n4) {
  int i = blockIdx.x * blockDim.x + threadIdx.x;
  const int stride = gridDim.x * blockDim.x;
  const float4*  in4  = (const float4*)in;
  unsigned int*  out4 = (unsigned int*)out;
  for (; i < n4; i += stride) {
    float4 f = in4[i];
    out4[i] = quant4(f.x, f.y, f.z, f.w);
  }
}

template <bool RELU_SQ_FP8>
__global__ void __launch_bounds__(256)
gemm_fp8_wmma_kernel(const unsigned char* __restrict__ A,
                     const unsigned char* __restrict__ B,
                     unsigned char* __restrict__ out8,
                     float* __restrict__ outf,
                     int M, int N, int K) {
  __shared__ unsigned char lA[BM][BK];
  __shared__ unsigned char lB[BN][BK];

  const int tid  = threadIdx.x;
  const int lane = tid & 31;
  const int wave = tid >> 5;
  const int wm   = wave & 3;
  const int wn   = wave >> 2;
  const int h    = lane >> 4;
  const int lr   = lane & 15;

  const long long m0 = (long long)blockIdx.x * BM;
  const long long n0 = (long long)blockIdx.y * BN;

  v8f acc[2][4];
  #pragma unroll
  for (int mt = 0; mt < 2; ++mt)
    #pragma unroll
    for (int nt = 0; nt < 4; ++nt) acc[mt][nt] = {};

  for (int k0 = 0; k0 < K; k0 += BK) {
    #pragma unroll
    for (int j = 0; j < 4; ++j) {
      int idx = tid + j * 256;
      int r   = idx >> 3;
      int c   = (idx & 7) << 4;
      copy16_to_lds(&A[(m0 + r) * (long long)K + k0 + c], &lA[r][c]);
      copy16_to_lds(&B[(n0 + r) * (long long)K + k0 + c], &lB[r][c]);
    }
    if (k0 + BK < K) {
      __builtin_prefetch(&A[(m0 + (tid >> 1)) * (long long)K + k0 + BK + (tid & 1) * 64], 0, 3);
      __builtin_prefetch(&B[(n0 + (tid >> 1)) * (long long)K + k0 + BK + (tid & 1) * 64], 0, 3);
    }
    wait_lds_copies();
    __syncthreads();

    union { v16i v; unsigned long long q[8]; } afr[2];
    #pragma unroll
    for (int mt = 0; mt < 2; ++mt) {
      const int arow = wm * 32 + mt * 16 + lr;
      #pragma unroll
      for (int j = 0; j < 8; ++j) {
        int off = ((j & 3) << 4) + ((j >> 2) << 6) + (h << 3);
        afr[mt].q[j] = *(const unsigned long long*)&lA[arow][off];
      }
    }
    #pragma unroll
    for (int nt = 0; nt < 4; ++nt) {
      union { v16i v; uint2 x[8]; } bfr;
      const int brow = wn * 64 + nt * 16 + lr;
      #pragma unroll
      for (int g = 0; g < 8; ++g)
        bfr.x[g] = *(const uint2*)&lB[brow][((g & 3) << 4) + ((g >> 2) << 6) + (h << 3)];
      #pragma unroll
      for (int mt = 0; mt < 2; ++mt)
        acc[mt][nt] = __builtin_amdgcn_wmma_f32_16x16x128_fp8_fp8(
            afr[mt].v, bfr.v, (short)0, acc[mt][nt], false, false);
    }
    __syncthreads();
  }

  #pragma unroll
  for (int mt = 0; mt < 2; ++mt) {
    #pragma unroll
    for (int nt = 0; nt < 4; ++nt) {
      #pragma unroll
      for (int r = 0; r < 8; ++r) {
        long long row = m0 + wm * 32 + mt * 16 + r + 8 * h;
        long long col = n0 + wn * 64 + nt * 16 + lr;
        float v = acc[mt][nt][r];
        if (RELU_SQ_FP8) {
          float sq = v > 0.0f ? v * v : 0.0f;
          out8[row * (long long)N + col] = quant1(sq);
        } else {
          outf[row * (long long)N + col] = v;
        }
      }
    }
  }
}

extern "C" void kernel_launch(void* const* d_in, const int* in_sizes, int n_in,
                              void* d_out, int out_size, void* d_ws, size_t ws_size,
                              hipStream_t stream) {
  (void)in_sizes; (void)n_in; (void)out_size; (void)ws_size;

  const float* x      = (const float*)d_in[0];
  const float* fc_w   = (const float*)d_in[1];
  const float* proj_w = (const float*)d_in[2];

  const int Mtot = 2 * 4096;
  const int D    = 1024;
  const int H    = 4096;

  unsigned char* x8 = (unsigned char*)d_ws;
  unsigned char* w8 = x8 + (size_t)Mtot * D;
  unsigned char* p8 = w8 + (size_t)H * D;
  unsigned char* h8 = p8 + (size_t)D * H;

  quantize_fp8_kernel<<<4096, 256, 0, stream>>>(x, x8, Mtot * D / 4);
  quantize_fp8_kernel<<<2048, 256, 0, stream>>>(fc_w, w8, H * D / 4);
  quantize_fp8_kernel<<<2048, 256, 0, stream>>>(proj_w, p8, D * H / 4);

  {
    dim3 grid(Mtot / BM, H / BN);
    gemm_fp8_wmma_kernel<true><<<grid, 256, 0, stream>>>(x8, w8, h8, nullptr,
                                                         Mtot, H, D);
  }
  {
    dim3 grid(Mtot / BM, D / BN);
    gemm_fp8_wmma_kernel<false><<<grid, 256, 0, stream>>>(h8, p8, nullptr,
                                                          (float*)d_out,
                                                          Mtot, D, H);
  }
}
